// SimpleMultiAgentValueModuleGCN_41214506172576
// MI455X (gfx1250) — matrix operand order repaired and checked by reading, not compiled, not run on hardware
//
#include <hip/hip_runtime.h>
#include <hip/hip_bf16.h>

typedef __attribute__((ext_vector_type(16))) __bf16 v16bf;
typedef __attribute__((ext_vector_type(8)))  float  v8f;

#define NUM_ENVS   4096
#define NUM_AGENTS 32
#define IN_DIM     32
#define GCN_DIM    64
#define RNN_DIM    64
#define E_PER      256

// bf16 weight workspace layout (element offsets):
//   [0,     2048)  WpreT[n*32+k]  = W_pre[k][n]   (64x32, Bt layout)
//   [2048,  6144)  WgcnT[n*64+k]  = W_gcn[k][n]   (64x64, Bt layout)
//   [6144, 18432)  Wih  [n*64+k]  = W_ih[n][k]    (192x64, already Bt)
//   [18432,30720)  Whh  [n*64+k]  = W_hh[n][k]    (192x64, already Bt)
#define WS_WPRET 0
#define WS_WGCNT 2048
#define WS_WIH   6144
#define WS_WHH   18432
#define WS_ELEMS 30720

__device__ __forceinline__ v8f wmma_bf16(v16bf a, v16bf b, v8f c) {
    // D = A(16x32 bf16) * B(32x16 bf16) + C(16x16 f32)
    return __builtin_amdgcn_wmma_f32_16x16x32_bf16(
        /*neg_a=*/false, a, /*neg_b=*/false, b,
        /*c_mod=*/(short)0, c, /*reuse_a=*/false, /*reuse_b=*/false);
}

// A-layout load from fp32 row base. K = koff+kb..kb+15 (contiguous).
__device__ __forceinline__ v16bf loadA_g(const float* __restrict__ rowp, int kb, int koff) {
    v16bf a;
    const float* q0 = rowp + koff + kb;
    const float* q1 = rowp + koff + kb + 8;
#pragma unroll
    for (int j = 0; j < 8; ++j) a[j] = (__bf16)q0[j];
#pragma unroll
    for (int j = 0; j < 8; ++j) a[8 + j] = (__bf16)q1[j];
    return a;
}

__device__ __forceinline__ v16bf loadA_bf(const __bf16* rowp, int kb, int koff) {
    v16bf a;
    const __bf16* q0 = rowp + koff + kb;
    const __bf16* q1 = rowp + koff + kb + 8;
#pragma unroll
    for (int j = 0; j < 8; ++j) a[j] = q0[j];
#pragma unroll
    for (int j = 0; j < 8; ++j) a[8 + j] = q1[j];
    return a;
}

// B-layout load, Bt stored row-major with K contiguous (fp32 source).
__device__ __forceinline__ v16bf loadB_row(const float* __restrict__ rowp, int kb16, int koff) {
    v16bf b;
    const float* q = rowp + koff + kb16;
#pragma unroll
    for (int j = 0; j < 16; ++j) b[j] = (__bf16)q[j];
    return b;
}

// B-layout load, Bt stored row-major with K contiguous (bf16 source: LDS or global ws).
__device__ __forceinline__ v16bf loadB_bf(const __bf16* rowp, int kb16, int koff) {
    v16bf b;
    const __bf16* q = rowp + koff + kb16;
#pragma unroll
    for (int j = 0; j < 16; ++j) b[j] = q[j];
    return b;
}

// B-layout load from k-major fp32 global W[k][n] (fallback path, strided).
__device__ __forceinline__ v16bf loadB_kmaj(const float* __restrict__ W, int ldn, int n,
                                            int kb16, int koff) {
    v16bf b;
#pragma unroll
    for (int j = 0; j < 16; ++j) b[j] = (__bf16)W[(size_t)(koff + kb16 + j) * ldn + n];
    return b;
}

__device__ __forceinline__ float sigmoidf_(float x) { return 1.0f / (1.0f + __expf(-x)); }

// ---- one-shot fp32 -> bf16 weight prep (stream-ordered before main kernel) ----
__global__ __launch_bounds__(256) void prep_weights_kernel(
    const float* __restrict__ W_pre, const float* __restrict__ W_gcn,
    const float* __restrict__ W_ih,  const float* __restrict__ W_hh,
    __bf16* __restrict__ ws)
{
    int i = blockIdx.x * 256 + threadIdx.x;
    if (i < WS_WGCNT) {                       // WpreT: n=i>>5, k=i&31
        int n = i >> 5, k = i & 31;
        ws[i] = (__bf16)W_pre[k * GCN_DIM + n];
    } else if (i < WS_WIH) {                  // WgcnT
        int j = i - WS_WGCNT;
        int n = j >> 6, k = j & 63;
        ws[i] = (__bf16)W_gcn[k * GCN_DIM + n];
    } else if (i < WS_WHH) {                  // Wih copy
        ws[i] = (__bf16)W_ih[i - WS_WIH];
    } else if (i < WS_ELEMS) {                // Whh copy
        ws[i] = (__bf16)W_hh[i - WS_WHH];
    }
}

template <bool PREP>
__global__ __launch_bounds__(128) void gcn_gru_value_kernel(
    const float* __restrict__ x,         // [4096,32,32]
    const long long* __restrict__ edge,  // [4096,2,256] int64
    const float* __restrict__ h0,        // [4096,32,64]
    const float* __restrict__ W_pre,     // [32,64]  (k-major)
    const float* __restrict__ b_pre,     // [64]
    const float* __restrict__ W_gcn,     // [64,64]  (k-major)
    const float* __restrict__ b_gcn,     // [64]
    const float* __restrict__ W_ih,      // [192,64]
    const float* __restrict__ W_hh,      // [192,64]
    const float* __restrict__ b_ih,      // [192]
    const float* __restrict__ b_hh,      // [192]
    const float* __restrict__ W_out,     // [2048]
    const float* __restrict__ b_out,     // [1]
    const __bf16* __restrict__ wsbf,     // bf16 weights (PREP) or nullptr
    float* __restrict__ out)             // [4096] value ++ [4096*2048] hidden
{
    __shared__ unsigned int sCnt[32 * 32];  // adjacency counts [col][row]
    __shared__ float  sDinv[32];
    __shared__ __bf16 sM32[32 * 32];        // normalized adjacency, row-major [c][r]
    __shared__ __bf16 sXf[32 * 64];         // xf, row-major (A layout)
    __shared__ __bf16 sHT[64 * 32];         // h transposed: hT[n][m] (Bt layout)
    __shared__ __bf16 sAgg[32 * 64];        // agg, row-major (A layout)
    __shared__ float  sAcc;

    const int env  = blockIdx.x;
    const int tid  = threadIdx.x;
    const int wave = tid >> 5;
    const int lane = tid & 31;
    const int ln   = lane & 15;
    const int hi   = lane >> 4;      // which half of the wave
    const int kbA  = hi * 16;        // A-layout K sub-offset
    const int kbB  = hi * 16;        // B-layout K sub-offset
    const int mC   = hi * 8;         // C-layout row offset

    // ---- init LDS ----
    if (tid == 0) sAcc = 0.0f;
#pragma unroll
    for (int i = 0; i < 8; ++i) sCnt[tid * 8 + i] = 0u;
    __syncthreads();

    // ---- edge histogram: counts[col][row] ----
    const long long* er = edge + (size_t)env * 2 * E_PER;
#pragma unroll
    for (int i = 0; i < 2; ++i) {
        int e = tid + i * 128;
        int r = (int)er[e];
        int c = (int)er[E_PER + e];
        atomicAdd(&sCnt[c * 32 + r], 1u);
    }

    // ---- stage 1: xf = x @ W_pre + b_pre (8 tiles, K=32) ----
    for (int t = wave; t < 8; t += 4) {
        int mblk = t >> 2, nblk = t & 3;
        int M = mblk * 16 + ln;
        int n = nblk * 16 + ln;
        v16bf a = loadA_g(x + (size_t)env * NUM_AGENTS * IN_DIM + M * IN_DIM, kbA, 0);
        v16bf b = PREP ? loadB_bf(wsbf + WS_WPRET + n * 32, kbB, 0)
                       : loadB_kmaj(W_pre, GCN_DIM, n, kbB, 0);
        v8f c = {};
        c = wmma_bf16(a, b, c);
        float bp = b_pre[n];
#pragma unroll
        for (int v = 0; v < 8; ++v) {
            int m = mblk * 16 + mC + v;
            sXf[m * 64 + n] = (__bf16)(c[v] + bp);
        }
    }
    __syncthreads();  // counts + xf ready

    // ---- degree / dinv (wave 0 only; full-wave uniform branch) ----
    if (tid < 32) {
        unsigned s = 0;
#pragma unroll
        for (int r = 0; r < 32; ++r) s += sCnt[tid * 32 + r];
        sDinv[tid] = rsqrtf((float)(s + 1u));  // +1 self loop
    }

    // ---- stage 2: h = xf @ W_gcn (8 tiles, K=64), store transposed ----
    for (int t = wave; t < 8; t += 4) {
        int mblk = t >> 2, nblk = t & 3;
        int M = mblk * 16 + ln;
        int n = nblk * 16 + ln;
        v8f c = {};
        if (PREP) {
            c = wmma_bf16(loadA_bf(sXf + M * 64, kbA, 0),
                          loadB_bf(wsbf + WS_WGCNT + n * 64, kbB, 0),  c);
            c = wmma_bf16(loadA_bf(sXf + M * 64, kbA, 32),
                          loadB_bf(wsbf + WS_WGCNT + n * 64, kbB, 32), c);
        } else {
            c = wmma_bf16(loadA_bf(sXf + M * 64, kbA, 0),
                          loadB_kmaj(W_gcn, GCN_DIM, n, kbB, 0),  c);
            c = wmma_bf16(loadA_bf(sXf + M * 64, kbA, 32),
                          loadB_kmaj(W_gcn, GCN_DIM, n, kbB, 32), c);
        }
#pragma unroll
        for (int v = 0; v < 8; ++v) {
            int m = mblk * 16 + mC + v;
            sHT[n * 32 + m] = (__bf16)c[v];
        }
    }
    __syncthreads();  // dinv + hT ready

    // ---- build normalized adjacency (A matrix of stage 3) ----
#pragma unroll
    for (int i = 0; i < 8; ++i) {
        int idx = tid * 8 + i;
        int c = idx >> 5, r = idx & 31;
        float cnt = (float)(sCnt[idx] + ((c == r) ? 1u : 0u));
        sM32[idx] = (__bf16)(cnt * sDinv[c] * sDinv[r]);
    }
    __syncthreads();

    // ---- stage 3: agg = M32 @ h + b_gcn (8 tiles, K=32) ----
    for (int t = wave; t < 8; t += 4) {
        int mblk = t >> 2, nblk = t & 3;
        int M = mblk * 16 + ln;
        int n = nblk * 16 + ln;
        v8f c = {};
        c = wmma_bf16(loadA_bf(sM32 + M * 32, kbA, 0), loadB_bf(sHT + n * 32, kbB, 0), c);
        float bg = b_gcn[n];
#pragma unroll
        for (int v = 0; v < 8; ++v) {
            int m = mblk * 16 + mC + v;
            sAgg[m * 64 + n] = (__bf16)(c[v] + bg);
        }
    }
    __syncthreads();

    // ---- stage 4: GRU gates + hnew + value partials ----
    float acc = 0.0f;
    const float* h0env = h0 + (size_t)env * NUM_AGENTS * RNN_DIM;
    for (int t = wave; t < 8; t += 4) {
        int mblk = t >> 2, dblk = t & 3;
        int M = mblk * 16 + ln;
        v16bf aAgg0 = loadA_bf(sAgg + M * 64, kbA, 0);
        v16bf aAgg1 = loadA_bf(sAgg + M * 64, kbA, 32);
        v16bf aH0   = loadA_g(h0env + M * 64, kbA, 0);
        v16bf aH1   = loadA_g(h0env + M * 64, kbA, 32);

        v8f ci[3], ch[3];
        float bi[3], bh[3];
#pragma unroll
        for (int g = 0; g < 3; ++g) {
            int nrow = g * 64 + dblk * 16 + ln;  // gate row of W_ih / W_hh
            v8f c = {};
            v8f d = {};
            if (PREP) {
                const __bf16* wi = wsbf + WS_WIH + nrow * 64;
                const __bf16* wh = wsbf + WS_WHH + nrow * 64;
                c = wmma_bf16(aAgg0, loadB_bf(wi, kbB, 0),  c);
                c = wmma_bf16(aAgg1, loadB_bf(wi, kbB, 32), c);
                d = wmma_bf16(aH0,   loadB_bf(wh, kbB, 0),  d);
                d = wmma_bf16(aH1,   loadB_bf(wh, kbB, 32), d);
            } else {
                c = wmma_bf16(aAgg0, loadB_row(W_ih + nrow * 64, kbB, 0),  c);
                c = wmma_bf16(aAgg1, loadB_row(W_ih + nrow * 64, kbB, 32), c);
                d = wmma_bf16(aH0,   loadB_row(W_hh + nrow * 64, kbB, 0),  d);
                d = wmma_bf16(aH1,   loadB_row(W_hh + nrow * 64, kbB, 32), d);
            }
            ci[g] = c;
            ch[g] = d;
            bi[g] = b_ih[nrow];
            bh[g] = b_hh[nrow];
        }

        int dcol = dblk * 16 + ln;
#pragma unroll
        for (int v = 0; v < 8; ++v) {
            int m = mblk * 16 + mC + v;
            float h0v = h0env[m * 64 + dcol];
            float rg = sigmoidf_(ci[0][v] + bi[0] + ch[0][v] + bh[0]);
            float zg = sigmoidf_(ci[1][v] + bi[1] + ch[1][v] + bh[1]);
            float ng = tanhf(ci[2][v] + bi[2] + rg * (ch[2][v] + bh[2]));
            float hn = (1.0f - zg) * ng + zg * h0v;
            out[NUM_ENVS + (size_t)env * (NUM_AGENTS * RNN_DIM) + m * 64 + dcol] = hn;
            acc += hn * W_out[m * 64 + dcol];
        }
    }

    // ---- value reduction (wave32 shuffle, then one LDS atomic per wave) ----
#pragma unroll
    for (int off = 16; off > 0; off >>= 1) acc += __shfl_down(acc, off);
    if (lane == 0) atomicAdd(&sAcc, acc);
    __syncthreads();
    if (tid == 0) out[env] = sAcc + b_out[0];
}

extern "C" void kernel_launch(void* const* d_in, const int* in_sizes, int n_in,
                              void* d_out, int out_size, void* d_ws, size_t ws_size,
                              hipStream_t stream) {
    const float*     x     = (const float*)d_in[0];
    const long long* edge  = (const long long*)d_in[1];
    const float*     h0    = (const float*)d_in[2];
    const float*     W_pre = (const float*)d_in[3];
    const float*     b_pre = (const float*)d_in[4];
    const float*     W_gcn = (const float*)d_in[5];
    const float*     b_gcn = (const float*)d_in[6];
    const float*     W_ih  = (const float*)d_in[7];
    const float*     W_hh  = (const float*)d_in[8];
    const float*     b_ih  = (const float*)d_in[9];
    const float*     b_hh  = (const float*)d_in[10];
    const float*     W_out = (const float*)d_in[11];
    const float*     b_out = (const float*)d_in[12];
    float* out = (float*)d_out;

    const bool prep = (ws_size >= (size_t)WS_ELEMS * sizeof(__bf16)) && (d_ws != nullptr);
    if (prep) {
        __bf16* wsbf = (__bf16*)d_ws;
        prep_weights_kernel<<<dim3((WS_ELEMS + 255) / 256), dim3(256), 0, stream>>>(
            W_pre, W_gcn, W_ih, W_hh, wsbf);
        gcn_gru_value_kernel<true><<<dim3(NUM_ENVS), dim3(128), 0, stream>>>(
            x, edge, h0, W_pre, b_pre, W_gcn, b_gcn,
            W_ih, W_hh, b_ih, b_hh, W_out, b_out, wsbf, out);
    } else {
        gcn_gru_value_kernel<false><<<dim3(NUM_ENVS), dim3(128), 0, stream>>>(
            x, edge, h0, W_pre, b_pre, W_gcn, b_gcn,
            W_ih, W_hh, b_ih, b_hh, W_out, b_out, nullptr, out);
    }
}
